// Aux_Free_Moe_48223892800347
// MI455X (gfx1250) — matrix operand order repaired and checked by reading, not compiled, not run on hardware
//
#include <hip/hip_runtime.h>
#include <hip/hip_bf16.h>

// ---------------------------------------------------------------------------
// MoE forward (top-2 of 8 experts), gathered formulation, bf16 WMMA compute.
//   B=8192 tokens, F=2048, H=8192, O=2048, E=8, TOP_K=2
// Pipeline: zero -> router(gather) -> per expert: GEMM1 (X->h, relu, bf16)
//           then GEMM2 (h->out, gate-scaled scatter accumulate).
// GEMMs: 256 thr = 8 waves, block tile 128x128, wave tile 64x32,
//        K-step 32, double-buffered LDS, 2-deep pipeline.
// GEMM2 A tile (bf16->bf16) uses GLOBAL_LOAD_ASYNC_TO_LDS_B128 (ASYNCcnt).
// Workspace: [0,32) cnt | [256,+256K) tok | [+256K) w | [524544,+128MB) h bf16
// ---------------------------------------------------------------------------

#define NE 8
#define NB 8192
#define NF 2048
#define NH 8192
#define NO 2048

typedef __bf16 bf16;
typedef __attribute__((ext_vector_type(16))) __bf16 bf16x16;
typedef __attribute__((ext_vector_type(8)))  float   f32x8;

union FragAB { bf16x16 v; uint4 q[2]; };

#define AS_STRIDE 40   // 32 K + 8 pad  (80B rows; 16B aligned, bank-spread)
#define BS_STRIDE 136  // 128 N + 8 pad (272B rows; 16B aligned, bank-spread)

// convert 16 fp32 (4 float4) -> 16 bf16 packed in two uint4
__device__ inline void cvt16_bf16(const float4* v, uint4& q0, uint4& q1) {
  bf16 t[16];
  t[0] = (bf16)v[0].x; t[1] = (bf16)v[0].y; t[2]  = (bf16)v[0].z; t[3]  = (bf16)v[0].w;
  t[4] = (bf16)v[1].x; t[5] = (bf16)v[1].y; t[6]  = (bf16)v[1].z; t[7]  = (bf16)v[1].w;
  t[8] = (bf16)v[2].x; t[9] = (bf16)v[2].y; t[10] = (bf16)v[2].z; t[11] = (bf16)v[2].w;
  t[12]= (bf16)v[3].x; t[13]= (bf16)v[3].y; t[14] = (bf16)v[3].z; t[15] = (bf16)v[3].w;
  q0 = ((const uint4*)t)[0];
  q1 = ((const uint4*)t)[1];
}

// async copy 32B global -> LDS (per lane), tracked by ASYNCcnt
__device__ inline void async_copy32(unsigned lds_addr, const void* gptr) {
  unsigned long long ga = (unsigned long long)gptr;
  asm volatile(
      "global_load_async_to_lds_b128 %0, %1, off\n\t"
      "global_load_async_to_lds_b128 %0, %1, off offset:16"
      :: "v"(lds_addr), "v"(ga) : "memory");
}

__device__ inline void wait_async0() {
#if __has_builtin(__builtin_amdgcn_s_wait_asynccnt)
  __builtin_amdgcn_s_wait_asynccnt(0);
#else
  asm volatile("s_wait_asynccnt 0x0" ::: "memory");
#endif
}

// ---------------------------------------------------------------- zero init
__global__ __launch_bounds__(256) void moe_zero(float* __restrict__ out,
                                                int* __restrict__ cnt) {
  size_t i  = (size_t)blockIdx.x * blockDim.x + threadIdx.x;
  size_t n4 = (size_t)NB * NO / 4;
  float4 z; z.x = z.y = z.z = z.w = 0.0f;
  for (size_t j = i; j < n4; j += (size_t)gridDim.x * blockDim.x)
    ((float4*)out)[j] = z;
  if (i < NE) cnt[i] = 0;
}

// ------------------------------------------------------------------- router
__global__ __launch_bounds__(256)
void moe_router(const float* __restrict__ X,  const float* __restrict__ gW,
                const float* __restrict__ gb, const float* __restrict__ eb,
                int* __restrict__ cnt, int* __restrict__ tok,
                float* __restrict__ wl) {
  const int wave = threadIdx.x >> 5;
  const int lane = threadIdx.x & 31;
  const int t    = blockIdx.x * 8 + wave;

  float acc[NE];
#pragma unroll
  for (int i = 0; i < NE; ++i) acc[i] = 0.0f;

  const float* x = X + (size_t)t * NF;
  for (int f = lane; f < NF; f += 32) {
    float xv = x[f];
    const float4* w = (const float4*)(gW + (size_t)f * NE);
    float4 w0 = w[0], w1 = w[1];
    acc[0] += xv * w0.x; acc[1] += xv * w0.y;
    acc[2] += xv * w0.z; acc[3] += xv * w0.w;
    acc[4] += xv * w1.x; acc[5] += xv * w1.y;
    acc[6] += xv * w1.z; acc[7] += xv * w1.w;
  }
#pragma unroll
  for (int off = 16; off > 0; off >>= 1)
#pragma unroll
    for (int i = 0; i < NE; ++i) acc[i] += __shfl_xor(acc[i], off, 32);

  if (lane == 0) {
    float s[NE];
#pragma unroll
    for (int i = 0; i < NE; ++i) s[i] = acc[i] + gb[i] + eb[i];
    int i0 = 0;
#pragma unroll
    for (int i = 1; i < NE; ++i) if (s[i] > s[i0]) i0 = i;   // ties -> low idx
    int i1 = (i0 == 0) ? 1 : 0;
#pragma unroll
    for (int i = 0; i < NE; ++i) if (i != i0 && s[i] > s[i1]) i1 = i;
    float e1 = __expf(s[i1] - s[i0]);     // s[i0] is the max; e0 = 1
    float inv = 1.0f / (1.0f + e1);
    int p0 = atomicAdd(&cnt[i0], 1);
    tok[i0 * NB + p0] = t; wl[i0 * NB + p0] = inv;
    int p1 = atomicAdd(&cnt[i1], 1);
    tok[i1 * NB + p1] = t; wl[i1 * NB + p1] = e1 * inv;
  }
}

// ------------------------------------------------------------ GEMM1 (X->h)
#define GEMM1_EPI(GUARD)                                                     \
  _Pragma("unroll") for (int mt = 0; mt < 4; ++mt) {                         \
    _Pragma("unroll") for (int nt = 0; nt < 2; ++nt) {                       \
      int n = n0 + wn * 32 + nt * 16 + col;                                  \
      float bias = b1[(size_t)e * NH + n];                                   \
      _Pragma("unroll") for (int i = 0; i < 8; ++i) {                        \
        int m = m0 + wm * 64 + mt * 16 + half * 8 + i;                       \
        if (GUARD) {                                                         \
          float v = acc[mt][nt][i] + bias;                                   \
          h[(size_t)m * NH + n] = (bf16)(v > 0.0f ? v : 0.0f);               \
        }                                                                    \
      }                                                                      \
    }                                                                        \
  }

__global__ __launch_bounds__(256)
void moe_gemm1(const float* __restrict__ X, const float* __restrict__ W1,
               const float* __restrict__ b1, const int* __restrict__ cnt,
               const int* __restrict__ tok, bf16* __restrict__ h, int e) {
  const int count = cnt[e];
  const int m0 = blockIdx.y * 128;
  if (m0 >= count) return;
  const int n0 = blockIdx.x * 128;         // over H
  const float* W = W1 + (size_t)e * NF * NH;

  const int tid  = threadIdx.x;
  const int lane = tid & 31, wave = tid >> 5;
  const int wm = wave >> 2, wn = wave & 3;

  __shared__ __align__(16) bf16 As[2][128 * AS_STRIDE];
  __shared__ __align__(16) bf16 Bs[2][32 * BS_STRIDE];

  f32x8 acc[4][2];
  const f32x8 fz = {0.f, 0.f, 0.f, 0.f, 0.f, 0.f, 0.f, 0.f};
#pragma unroll
  for (int mt = 0; mt < 4; ++mt)
#pragma unroll
    for (int nt = 0; nt < 2; ++nt) acc[mt][nt] = fz;

  const int arow = tid >> 1, ahalf = tid & 1;        // A: 2 thr / row
  const int am   = m0 + arow;
  const int atok = (am < count) ? tok[e * NB + am] : 0;
  const float* aptr = X + (size_t)atok * NF + ahalf * 16;
  const int brow = tid >> 3, bseg = tid & 7;         // B: 8 thr / row
  const float* bptr = W + (size_t)brow * NH + n0 + bseg * 16;

  float4 ra[4], rb[4];
  { // prologue: tile k0 = 0
    const float4* pa = (const float4*)aptr;
    ra[0] = pa[0]; ra[1] = pa[1]; ra[2] = pa[2]; ra[3] = pa[3];
    const float4* pb = (const float4*)bptr;
    rb[0] = pb[0]; rb[1] = pb[1]; rb[2] = pb[2]; rb[3] = pb[3];
    uint4 q0, q1;
    cvt16_bf16(ra, q0, q1);
    uint4* da = (uint4*)&As[0][arow * AS_STRIDE + ahalf * 16];
    da[0] = q0; da[1] = q1;
    cvt16_bf16(rb, q0, q1);
    uint4* db = (uint4*)&Bs[0][brow * BS_STRIDE + bseg * 16];
    db[0] = q0; db[1] = q1;
  }
  __syncthreads();

  int cur = 0;
  for (int k0 = 0; k0 < NF; k0 += 32) {
    const bool more = (k0 + 32) < NF;
    if (more) { // fetch tile k0+32 into registers while computing
      const float4* pa = (const float4*)(aptr + k0 + 32);
      ra[0] = pa[0]; ra[1] = pa[1]; ra[2] = pa[2]; ra[3] = pa[3];
      const float4* pb = (const float4*)(bptr + (size_t)(k0 + 32) * NH);
      rb[0] = pb[0]; rb[1] = pb[1]; rb[2] = pb[2]; rb[3] = pb[3];
      if (k0 + 64 < NF)  // warm GL2 for the tile after next (weights dominate)
        __builtin_prefetch(bptr + (size_t)(k0 + 64) * NH, 0, 1);
    }

    FragAB a[4], b[2];
#pragma unroll
    for (int mt = 0; mt < 4; ++mt) {
      const bf16* s =
          &As[cur][(wm * 64 + mt * 16 + (lane & 15)) * AS_STRIDE + (lane >> 4) * 8];
      a[mt].q[0] = *(const uint4*)s;
      a[mt].q[1] = *(const uint4*)(s + 16);
    }
#pragma unroll
    for (int nt = 0; nt < 2; ++nt) {
      const bf16* s = &Bs[cur][(lane >> 4) * 8 * BS_STRIDE + wn * 32 + nt * 16 + (lane & 15)];
#pragma unroll
      for (int i = 0; i < 8; ++i) {
        b[nt].v[i] = s[i * BS_STRIDE];
        b[nt].v[8 + i] = s[(16 + i) * BS_STRIDE];
      }
    }
#pragma unroll
    for (int mt = 0; mt < 4; ++mt)
#pragma unroll
      for (int nt = 0; nt < 2; ++nt)
        acc[mt][nt] = __builtin_amdgcn_wmma_f32_16x16x32_bf16(
            false, a[mt].v, false, b[nt].v, (short)0, acc[mt][nt], false, false);

    if (more) { // store next tile into the alternate buffer
      uint4 q0, q1;
      cvt16_bf16(ra, q0, q1);
      uint4* da = (uint4*)&As[cur ^ 1][arow * AS_STRIDE + ahalf * 16];
      da[0] = q0; da[1] = q1;
      cvt16_bf16(rb, q0, q1);
      uint4* db = (uint4*)&Bs[cur ^ 1][brow * BS_STRIDE + bseg * 16];
      db[0] = q0; db[1] = q1;
    }
    __syncthreads();
    cur ^= 1;
  }

  const int col = lane & 15, half = lane >> 4;
  if (m0 + 128 <= count) {
    GEMM1_EPI(true)
  } else {
    GEMM1_EPI(m < count)
  }
}

// --------------------------------------------------------- GEMM2 (h->out)
#define GEMM2_EPI(GUARD)                                                     \
  _Pragma("unroll") for (int mt = 0; mt < 4; ++mt) {                         \
    _Pragma("unroll") for (int nt = 0; nt < 2; ++nt) {                       \
      int n = n0 + wn * 32 + nt * 16 + col;                                  \
      float bias = b2[(size_t)e * NO + n];                                   \
      _Pragma("unroll") for (int i = 0; i < 8; ++i) {                        \
        int m = m0 + wm * 64 + mt * 16 + half * 8 + i;                       \
        if (GUARD) {                                                         \
          int   t = tok[e * NB + m];                                         \
          float w = wl[e * NB + m];                                          \
          float* o = &out[(size_t)t * NO + n];                               \
          *o = *o + w * (acc[mt][nt][i] + bias);                             \
        }                                                                    \
      }                                                                      \
    }                                                                        \
  }

__global__ __launch_bounds__(256)
void moe_gemm2(const bf16* __restrict__ h, const float* __restrict__ W2,
               const float* __restrict__ b2, const int* __restrict__ cnt,
               const int* __restrict__ tok, const float* __restrict__ wl,
               float* __restrict__ out, int e) {
  const int count = cnt[e];
  const int m0 = blockIdx.y * 128;
  if (m0 >= count) return;
  const int n0 = blockIdx.x * 128;         // over O
  const float* W = W2 + (size_t)e * NH * NO;

  const int tid  = threadIdx.x;
  const int lane = tid & 31, wave = tid >> 5;
  const int wm = wave >> 2, wn = wave & 3;

  __shared__ __align__(16) bf16 As[2][128 * AS_STRIDE];
  __shared__ __align__(16) bf16 Bs[2][32 * BS_STRIDE];

  f32x8 acc[4][2];
  const f32x8 fz = {0.f, 0.f, 0.f, 0.f, 0.f, 0.f, 0.f, 0.f};
#pragma unroll
  for (int mt = 0; mt < 4; ++mt)
#pragma unroll
    for (int nt = 0; nt < 2; ++nt) acc[mt][nt] = fz;

  const int arow = tid >> 1, ahalf = tid & 1;
  const bf16* aptr = h + (size_t)(m0 + arow) * NH + ahalf * 16;
  const int brow = tid >> 3, bseg = tid & 7;
  const float* bptr = W + (size_t)brow * NO + n0 + bseg * 16;

  // LDS byte addresses for this thread's async A-tile slot (both buffers):
  // flat->LDS aperture mapping keeps the offset in addr[31:0].
  const unsigned ldsA0 =
      (unsigned)(uintptr_t)&As[0][arow * AS_STRIDE + ahalf * 16];
  const unsigned ldsA1 =
      (unsigned)(uintptr_t)&As[1][arow * AS_STRIDE + ahalf * 16];

  float4 rb[4];
  { // prologue: tile k0 = 0 (A via async copy engine, B via cvt path)
    async_copy32(ldsA0, aptr);
    const float4* pb = (const float4*)bptr;
    rb[0] = pb[0]; rb[1] = pb[1]; rb[2] = pb[2]; rb[3] = pb[3];
    uint4 q0, q1;
    cvt16_bf16(rb, q0, q1);
    uint4* db = (uint4*)&Bs[0][brow * BS_STRIDE + bseg * 16];
    db[0] = q0; db[1] = q1;
    wait_async0();
  }
  __syncthreads();

  int cur = 0;
  for (int k0 = 0; k0 < NH; k0 += 32) {
    const bool more = (k0 + 32) < NH;
    if (more) {
      // A tile k0+32: async global->LDS (overlaps with the WMMAs below)
      async_copy32(cur ? ldsA0 : ldsA1, aptr + k0 + 32);
      const float4* pb = (const float4*)(bptr + (size_t)(k0 + 32) * NO);
      rb[0] = pb[0]; rb[1] = pb[1]; rb[2] = pb[2]; rb[3] = pb[3];
      if (k0 + 64 < NH)
        __builtin_prefetch(bptr + (size_t)(k0 + 64) * NO, 0, 1);
    }

    FragAB a[4], b[2];
#pragma unroll
    for (int mt = 0; mt < 4; ++mt) {
      const bf16* s =
          &As[cur][(wm * 64 + mt * 16 + (lane & 15)) * AS_STRIDE + (lane >> 4) * 8];
      a[mt].q[0] = *(const uint4*)s;
      a[mt].q[1] = *(const uint4*)(s + 16);
    }
#pragma unroll
    for (int nt = 0; nt < 2; ++nt) {
      const bf16* s = &Bs[cur][(lane >> 4) * 8 * BS_STRIDE + wn * 32 + nt * 16 + (lane & 15)];
#pragma unroll
      for (int i = 0; i < 8; ++i) {
        b[nt].v[i] = s[i * BS_STRIDE];
        b[nt].v[8 + i] = s[(16 + i) * BS_STRIDE];
      }
    }
#pragma unroll
    for (int mt = 0; mt < 4; ++mt)
#pragma unroll
      for (int nt = 0; nt < 2; ++nt)
        acc[mt][nt] = __builtin_amdgcn_wmma_f32_16x16x32_bf16(
            false, a[mt].v, false, b[nt].v, (short)0, acc[mt][nt], false, false);

    if (more) {
      uint4 q0, q1;
      cvt16_bf16(rb, q0, q1);
      uint4* db = (uint4*)&Bs[cur ^ 1][brow * BS_STRIDE + bseg * 16];
      db[0] = q0; db[1] = q1;
      wait_async0();   // A tile for buffer cur^1 fully landed in LDS
    }
    __syncthreads();
    cur ^= 1;
  }

  const int col = lane & 15, half = lane >> 4;
  if (m0 + 128 <= count) {
    GEMM2_EPI(true)
  } else {
    GEMM2_EPI(m < count)
  }
}

// ---------------------------------------------------------------- launcher
extern "C" void kernel_launch(void* const* d_in, const int* in_sizes, int n_in,
                              void* d_out, int out_size, void* d_ws,
                              size_t ws_size, hipStream_t stream) {
  (void)in_sizes; (void)n_in; (void)out_size; (void)ws_size;
  const float* X  = (const float*)d_in[0];
  const float* gW = (const float*)d_in[1];
  const float* gb = (const float*)d_in[2];
  const float* eb = (const float*)d_in[3];
  const float* W1 = (const float*)d_in[4];
  const float* b1 = (const float*)d_in[5];
  const float* W2 = (const float*)d_in[6];
  const float* b2 = (const float*)d_in[7];
  float* out = (float*)d_out;

  char* ws = (char*)d_ws;
  int*   cnt = (int*)ws;
  int*   tok = (int*)(ws + 256);
  float* wl  = (float*)(ws + 256 + (size_t)NE * NB * 4);
  bf16*  h   = (bf16*)(ws + 256 + (size_t)NE * NB * 8);

  moe_zero<<<2048, 256, 0, stream>>>(out, cnt);
  moe_router<<<NB / 8, 256, 0, stream>>>(X, gW, gb, eb, cnt, tok, wl);
  for (int e = 0; e < NE; ++e) {
    moe_gemm1<<<dim3(NH / 128, NB / 128), 256, 0, stream>>>(X, W1, b1, cnt,
                                                            tok, h, e);
    moe_gemm2<<<dim3(NO / 128, NB / 128), 256, 0, stream>>>(h, W2, b2, cnt,
                                                            tok, wl, out, e);
  }
}
